// Attention_40664750358615
// MI455X (gfx1250) — compile-verified, matrix operand order checked by reading, not run on hardware
//
#include <hip/hip_runtime.h>

// ---------------------------------------------------------------------------
// MI455X (gfx1250) adapter-attention forward.
// bf16 WMMA (v_wmma_f32_16x16x32_bf16) for all GEMMs + flash attention.
// Attention K/V tiles staged by the Tensor Data Mover (tensor_load_to_lds).
// ---------------------------------------------------------------------------

#define DIM   4096
#define HEADS 32
#define HD    128
#define SEQ   2048
#define BSZ   2
#define ALEN  10
#define MROWS (BSZ * SEQ)   // 4096

typedef __attribute__((ext_vector_type(16))) __bf16 v16bf;
typedef __attribute__((ext_vector_type(8)))  float  v8f;
typedef __attribute__((ext_vector_type(4)))  unsigned int u32x4;
typedef __attribute__((ext_vector_type(8)))  unsigned int u32x8;

__device__ __forceinline__ unsigned short f2bf(float f) {
  union { float f; unsigned int u; } v; v.f = f;
  unsigned int r = v.u + 0x7FFFu + ((v.u >> 16) & 1u);
  return (unsigned short)(r >> 16);
}
__device__ __forceinline__ float bf2f(unsigned short b) {
  union { unsigned int u; float f; } v; v.u = ((unsigned int)b) << 16;
  return v.f;
}
__device__ __forceinline__ unsigned rfl(unsigned x) {
  return (unsigned)__builtin_amdgcn_readfirstlane((int)x);
}

union FragU { uint4 u[2]; v16bf v; };

// WMMA 16-lane-striped fragment: elems 0-7 at p[0..7], 8-15 at p[16..23]
// (caller bakes in the +8*(lane>=16) K-half offset). Both 16B loads aligned.
__device__ __forceinline__ v16bf ld_frag(const unsigned short* p) {
  FragU f;
  f.u[0] = *(const uint4*)(p);
  f.u[1] = *(const uint4*)(p + 16);
  return f.v;
}

// ---------------------------------------------------------------------------
// Tensor Data Mover: 2D tile (tile_w x tile_h elems, 2B each, row stride in
// elems) global -> LDS with per-row padding. Issued once per wave (uniform
// SGPR descriptors), tracked with TENSORcnt.
// pad codes: interval 0=2,1=4,2=8,3=16,4=32,5=64 DWORDs ; amount 3 = 4 DWORDs
// ---------------------------------------------------------------------------
__device__ __forceinline__ void tdm_load_2d(unsigned lds_addr, const void* gp,
                                            unsigned tile_w, unsigned tile_h,
                                            unsigned long long stride_elems,
                                            unsigned pad_int_code,
                                            unsigned pad_amt_code) {
  unsigned long long ga = (unsigned long long)gp;
  u32x4 g0;
  g0[0] = rfl(1u);                                   // count=1 (valid user D#)
  g0[1] = rfl(lds_addr);                             // lds_addr [63:32]
  g0[2] = rfl((unsigned)ga);                         // global_addr[31:0]
  g0[3] = rfl(((unsigned)(ga >> 32) & 0x01FFFFFFu) | (2u << 30)); // [56:32]|type=2
  u32x8 g1;
  g1[0] = rfl((1u << 16) | (1u << 20) |              // data_size=2B, pad_enable
              (pad_int_code << 22) | (pad_amt_code << 25));
  g1[1] = rfl(0xFFFF0000u);                          // tensor_dim0 lo16 (huge)
  g1[2] = rfl(0xFFFF7FFFu);                          // dim0 hi16 | dim1 lo16
  g1[3] = rfl(0x00007FFFu | (tile_w << 16));         // dim1 hi16 | tile_dim0
  g1[4] = rfl(tile_h & 0xFFFFu);                     // tile_dim1 ; tile_dim2=0
  g1[5] = rfl((unsigned)stride_elems);               // dim0_stride[31:0]
  g1[6] = rfl((unsigned)(stride_elems >> 32) & 0xFFFFu); // stride[47:32]
  g1[7] = rfl(0u);
  asm volatile("tensor_load_to_lds %0, %1" :: "s"(g0), "s"(g1) : "memory");
}

// ---------------------------------------------------------------------------
// elementwise f32 -> bf16
// ---------------------------------------------------------------------------
__global__ void cvt_f32_bf16(const float* __restrict__ in,
                             unsigned short* __restrict__ out, long long n) {
  long long i = (long long)blockIdx.x * blockDim.x + threadIdx.x;
  long long stride = (long long)gridDim.x * blockDim.x;
  for (; i < n; i += stride) out[i] = f2bf(in[i]);
}

// ---------------------------------------------------------------------------
// DIM x DIM weight transpose, f32 in -> bf16 out: WT[n][k] = W[k][n]
// ---------------------------------------------------------------------------
__global__ __launch_bounds__(256) void transpose_w(const float* __restrict__ W,
                                                   unsigned short* __restrict__ WT) {
  __shared__ unsigned short t[32][33];
  int c0 = blockIdx.x * 32, r0 = blockIdx.y * 32;
  int tx = threadIdx.x, ty = threadIdx.y;   // 32 x 8
  for (int i = 0; i < 4; i++)
    t[ty + i * 8][tx] = f2bf(W[(long long)(r0 + ty + i * 8) * DIM + c0 + tx]);
  __syncthreads();
  for (int i = 0; i < 4; i++)
    WT[(long long)(c0 + ty + i * 8) * DIM + r0 + tx] = t[tx][ty + i * 8];
}

// ---------------------------------------------------------------------------
// V transpose: Vbf[(b*SEQ+s)][h*HD+hd] -> VT[((b*H+h)*HD+hd)][s]   (bf16)
// ---------------------------------------------------------------------------
__global__ __launch_bounds__(256) void transpose_v(const unsigned short* __restrict__ V,
                                                   unsigned short* __restrict__ VT) {
  __shared__ unsigned short t[32][33];
  int s0 = blockIdx.x * 32;
  int hdtile = blockIdx.y & 3;
  int bh = blockIdx.y >> 2;                 // b*HEADS + h
  int b = bh >> 5, h = bh & 31;
  int tx = threadIdx.x, ty = threadIdx.y;   // 32 x 8
  for (int i = 0; i < 4; i++)
    t[ty + i * 8][tx] = V[(long long)(b * SEQ + s0 + ty + i * 8) * DIM +
                          h * HD + hdtile * 32 + tx];
  __syncthreads();
  for (int i = 0; i < 4; i++)
    VT[((long long)bh * HD + hdtile * 32 + ty + i * 8) * SEQ + s0 + tx] =
        t[tx][ty + i * 8];
}

// ---------------------------------------------------------------------------
// GEMM: C[M,N] = A[M,K] * BT[N,K]^T  (A, BT bf16 row-major; C f32 or bf16)
// Block tile 128x128, BK=32, 8 waves (4x2), each wave 32x64 = 2x4 WMMA tiles.
// ---------------------------------------------------------------------------
#define GM_LDA 40   // 32 + 8 pad (ushorts); 80B rows -> conflict-free 16B frags

template <bool OUTF32>
__global__ __launch_bounds__(256) void gemm_bf16(
    const unsigned short* __restrict__ A, const unsigned short* __restrict__ BT,
    float* __restrict__ Cf, unsigned short* __restrict__ Cb,
    int M, int N, int K) {
  __shared__ unsigned short As[128 * GM_LDA];
  __shared__ unsigned short Bs[128 * GM_LDA];

  const int tid  = threadIdx.x;
  const int lane = tid & 31;
  const int wid  = tid >> 5;
  const int wm   = wid >> 1;        // 0..3 -> 32 rows each
  const int wn   = wid & 1;         // 0..1 -> 64 cols each
  const int half = lane >> 4;
  const int l15  = lane & 15;
  const int m0 = blockIdx.y * 128;
  const int n0 = blockIdx.x * 128;

  v8f acc[2][4];
  for (int i = 0; i < 2; i++)
    for (int j = 0; j < 4; j++)
      for (int e = 0; e < 8; e++) acc[i][j][e] = 0.0f;

  const int row = tid >> 1;
  const int kk  = (tid & 1) * 16;
  for (int k0 = 0; k0 < K; k0 += 32) {
    { // A tile 128x32 (rows of A)
      uint4 a0, a1;
      if (m0 + row < M) {
        const unsigned short* g = A + (long long)(m0 + row) * K + k0 + kk;
        a0 = *(const uint4*)(g);
        a1 = *(const uint4*)(g + 8);
      } else {
        a0 = make_uint4(0u, 0u, 0u, 0u); a1 = a0;
      }
      *(uint4*)(&As[row * GM_LDA + kk])     = a0;
      *(uint4*)(&As[row * GM_LDA + kk + 8]) = a1;
    }
    { // B tile 128x32 (rows of BT = columns of B) -- contiguous, no scatter
      const unsigned short* g = BT + (long long)(n0 + row) * K + k0 + kk;
      *(uint4*)(&Bs[row * GM_LDA + kk])     = *(const uint4*)(g);
      *(uint4*)(&Bs[row * GM_LDA + kk + 8]) = *(const uint4*)(g + 8);
    }
    __syncthreads();

    v16bf af[2], bfr[4];
    for (int i = 0; i < 2; i++)
      af[i] = ld_frag(&As[(wm * 32 + i * 16 + l15) * GM_LDA + half * 8]);
    for (int j = 0; j < 4; j++)
      bfr[j] = ld_frag(&Bs[(wn * 64 + j * 16 + l15) * GM_LDA + half * 8]);

    for (int i = 0; i < 2; i++)
      for (int j = 0; j < 4; j++)
        acc[i][j] = __builtin_amdgcn_wmma_f32_16x16x32_bf16(
            false, af[i], false, bfr[j], (short)0, acc[i][j], false, false);
    __syncthreads();
  }

  for (int i = 0; i < 2; i++)
    for (int j = 0; j < 4; j++) {
      int col = n0 + wn * 64 + j * 16 + l15;
      for (int e = 0; e < 8; e++) {
        int r = m0 + wm * 32 + i * 16 + e + half * 8;
        if (r < M) {
          long long idx = (long long)r * N + col;
          if (OUTF32) Cf[idx] = acc[i][j][e];
          else        Cb[idx] = f2bf(acc[i][j][e]);
        }
      }
    }
}

// ---------------------------------------------------------------------------
// In-place RoPE on bf16 tensor laid out (B*S, H, HD)
// ---------------------------------------------------------------------------
__global__ void rope_bf16(unsigned short* __restrict__ T,
                          const float* __restrict__ cs,
                          const float* __restrict__ sn) {
  long long idx = (long long)blockIdx.x * blockDim.x + threadIdx.x;
  if (idx >= (long long)MROWS * HEADS * (HD / 2)) return;
  int i = (int)(idx & 63);
  int h = (int)((idx >> 6) & 31);
  long long bs = idx >> 11;              // b*SEQ + s
  int s = (int)(bs & (SEQ - 1));
  long long base = bs * DIM + (long long)h * HD + 2 * i;
  float tr = bf2f(T[base]), ti = bf2f(T[base + 1]);
  float c = cs[s * (HD / 2) + i], d = sn[s * (HD / 2) + i];
  T[base]     = f2bf(tr * c - ti * d);
  T[base + 1] = f2bf(tr * d + ti * c);
}

// ---------------------------------------------------------------------------
// Flash attention: block = (q-tile of 128, one (b,h)); 8 waves x 16 q-rows.
// Key tile 64. K tile [key][hd] and V tile [hd][key] staged by TDM (wave 0
// issues tensor_load_to_lds, waits TENSORcnt, then workgroup barrier).
// P staged through LDS to convert WMMA C-layout -> A-layout.
// ---------------------------------------------------------------------------
#define AT_KT   64
#define AT_LDK  (HD + 8)      // 136 ushorts: 256B row + 16B TDM pad
#define AT_LDV  (AT_KT + 8)   // 72 ushorts: 128B row + 16B TDM pad

__global__ __launch_bounds__(256) void attn_kernel(
    const unsigned short* __restrict__ Qb, const unsigned short* __restrict__ Kb,
    const unsigned short* __restrict__ VT, float* __restrict__ ctx) {
  __shared__ unsigned short Ks[AT_KT * AT_LDK];   // [key][hd]
  __shared__ unsigned short Vs[HD * AT_LDV];      // [hd][key]
  __shared__ unsigned short Ps[128 * AT_LDV];     // [q-row][key]

  const int tid  = threadIdx.x;
  const int lane = tid & 31;
  const int wid  = tid >> 5;
  const int half = lane >> 4;
  const int l15  = lane & 15;
  const int qtile = blockIdx.x;              // 0..15
  const int b = blockIdx.y >> 5;
  const int h = blockIdx.y & 31;
  const float scale = 0.08838834764831845f;  // 1/sqrt(128)
  const long long rowbase = (long long)b * SEQ;

  // Q fragments straight from global (read exactly once per block)
  v16bf qf[4];
  {
    long long qrow = rowbase + qtile * 128 + wid * 16 + l15;
    const unsigned short* g = Qb + qrow * DIM + h * HD;
    for (int ks = 0; ks < 4; ks++)
      qf[ks] = ld_frag(g + ks * 32 + half * 8);
  }

  v8f O[8];
  float mrow[8], lrow[8];
  for (int nt = 0; nt < 8; nt++)
    for (int e = 0; e < 8; e++) O[nt][e] = 0.0f;
  for (int e = 0; e < 8; e++) { mrow[e] = -1e30f; lrow[e] = 0.0f; }

  const int ktiles = 2 * qtile + 2;          // causal: keys <= last q of tile
  for (int kt = 0; kt < ktiles; kt++) {
    if (wid == 0) {  // TDM stages both tiles; one DMA op each
      unsigned ldsK = (unsigned)(unsigned long long)(void*)Ks;
      const unsigned short* gk =
          Kb + (rowbase + (long long)kt * AT_KT) * DIM + h * HD;
      tdm_load_2d(ldsK, gk, HD, AT_KT, (unsigned long long)DIM, 5u, 3u);
      unsigned ldsV = (unsigned)(unsigned long long)(void*)Vs;
      const unsigned short* gv =
          VT + ((long long)(b * HEADS + h) * HD) * SEQ + kt * AT_KT;
      tdm_load_2d(ldsV, gv, AT_KT, HD, (unsigned long long)SEQ, 4u, 3u);
      __builtin_amdgcn_s_wait_tensorcnt(0);
    }
    __syncthreads();

    // S = Q K^T  (16 q-rows x 64 keys per wave)
    v8f S[4];
    for (int nt = 0; nt < 4; nt++) {
      for (int e = 0; e < 8; e++) S[nt][e] = 0.0f;
      for (int ks = 0; ks < 4; ks++) {
        v16bf bf = ld_frag(&Ks[(nt * 16 + l15) * AT_LDK + ks * 32 + half * 8]);
        S[nt] = __builtin_amdgcn_wmma_f32_16x16x32_bf16(
            false, qf[ks], false, bf, (short)0, S[nt], false, false);
      }
    }

    // scale + causal mask + per-row tile max (C layout: row = e + 8*half)
    const int qg = qtile * 128 + wid * 16 + half * 8;  // + e
    const int kg = kt * AT_KT + l15;                   // + nt*16
    float tmax[8];
    for (int e = 0; e < 8; e++) tmax[e] = -1e30f;
    for (int nt = 0; nt < 4; nt++)
      for (int e = 0; e < 8; e++) {
        float s = S[nt][e] * scale;
        if (kg + nt * 16 > qg + e) s = -1e30f;
        S[nt][e] = s;
        tmax[e] = fmaxf(tmax[e], s);
      }
    for (int m = 1; m < 16; m <<= 1)
      for (int e = 0; e < 8; e++)
        tmax[e] = fmaxf(tmax[e], __shfl_xor(tmax[e], m, 32));

    float alpha[8];
    for (int e = 0; e < 8; e++) {
      float mn = fmaxf(mrow[e], tmax[e]);
      alpha[e] = __expf(mrow[e] - mn);
      mrow[e] = mn;
    }

    // P = exp(S - m): stage to LDS (A-layout reload), accumulate row sums
    float tsum[8];
    for (int e = 0; e < 8; e++) tsum[e] = 0.0f;
    for (int nt = 0; nt < 4; nt++)
      for (int e = 0; e < 8; e++) {
        float p = __expf(S[nt][e] - mrow[e]);
        tsum[e] += p;
        Ps[(wid * 16 + e + half * 8) * AT_LDV + nt * 16 + l15] = f2bf(p);
      }
    for (int m = 1; m < 16; m <<= 1)
      for (int e = 0; e < 8; e++) tsum[e] += __shfl_xor(tsum[e], m, 32);
    for (int e = 0; e < 8; e++) lrow[e] = lrow[e] * alpha[e] + tsum[e];
    for (int nt = 0; nt < 8; nt++)
      for (int e = 0; e < 8; e++) O[nt][e] *= alpha[e];
    __syncthreads();

    // O += P V
    for (int ks = 0; ks < 2; ks++) {
      v16bf pa = ld_frag(&Ps[(wid * 16 + l15) * AT_LDV + ks * 32 + half * 8]);
      for (int nt = 0; nt < 8; nt++) {
        v16bf vb = ld_frag(&Vs[(nt * 16 + l15) * AT_LDV + ks * 32 + half * 8]);
        O[nt] = __builtin_amdgcn_wmma_f32_16x16x32_bf16(
            false, pa, false, vb, (short)0, O[nt], false, false);
      }
    }
    __syncthreads();
  }

  // ctx = O / l  (f32; adapter term added by next kernel)
  for (int nt = 0; nt < 8; nt++) {
    int col = h * HD + nt * 16 + l15;
    for (int e = 0; e < 8; e++) {
      long long q = rowbase + qtile * 128 + wid * 16 + e + half * 8;
      ctx[q * DIM + col] = O[nt][e] / lrow[e];
    }
  }
  (void)lane;
}

// ---------------------------------------------------------------------------
// Adapter path: per (b,s,h): 10-way softmax(q.ak*scale) * tanh(gate) @ av
// ---------------------------------------------------------------------------
__global__ void adapter_kernel(const unsigned short* __restrict__ Qb,
                               const unsigned short* __restrict__ akb,
                               const unsigned short* __restrict__ avb,
                               const float* __restrict__ gate,
                               float* __restrict__ ctx) {
  long long idx = (long long)blockIdx.x * blockDim.x + threadIdx.x;
  if (idx >= (long long)MROWS * HEADS) return;
  int h = (int)(idx & 31);
  long long bs = idx >> 5;
  const float scale = 0.08838834764831845f;
  const unsigned short* q = Qb + bs * DIM + h * HD;

  float sc[ALEN];
  float mx = -1e30f;
  for (int l = 0; l < ALEN; l++) {
    const unsigned short* ak = akb + (long long)l * DIM + h * HD;
    float a = 0.0f;
    for (int d = 0; d < HD; d++) a += bf2f(q[d]) * bf2f(ak[d]);
    sc[l] = a * scale;
    mx = fmaxf(mx, sc[l]);
  }
  float sum = 0.0f;
  for (int l = 0; l < ALEN; l++) { sc[l] = __expf(sc[l] - mx); sum += sc[l]; }
  float g = tanhf(gate[h]) / sum;

  float* o = ctx + bs * DIM + h * HD;
  for (int d = 0; d < HD; d++) {
    float acc = 0.0f;
    for (int l = 0; l < ALEN; l++)
      acc += sc[l] * bf2f(avb[(long long)l * DIM + h * HD + d]);
    o[d] += g * acc;
  }
}

// ---------------------------------------------------------------------------
// Host orchestration
// ---------------------------------------------------------------------------
extern "C" void kernel_launch(void* const* d_in, const int* in_sizes, int n_in,
                              void* d_out, int out_size, void* d_ws, size_t ws_size,
                              hipStream_t stream) {
  (void)in_sizes; (void)n_in; (void)out_size; (void)ws_size;
  const float* x       = (const float*)d_in[0];
  const float* wq      = (const float*)d_in[1];
  const float* wk      = (const float*)d_in[2];
  const float* wv      = (const float*)d_in[3];
  const float* wo      = (const float*)d_in[4];
  const float* gate    = (const float*)d_in[5];
  const float* adapter = (const float*)d_in[6];
  const float* fcos    = (const float*)d_in[7];
  const float* fsin    = (const float*)d_in[8];
  // d_in[9]=mask (causal, applied analytically), [10]=start_pos(0), [11]=random_init(0)

  char* ws = (char*)d_ws;
  size_t off = 0;
  auto alloc = [&](size_t bytes) -> void* {
    off = (off + 255) & ~(size_t)255;
    void* p = ws + off;
    off += bytes;
    return p;
  };
  unsigned short* xb   = (unsigned short*)alloc((size_t)MROWS * DIM * 2);
  unsigned short* wqT  = (unsigned short*)alloc((size_t)DIM * DIM * 2);
  unsigned short* wkT  = (unsigned short*)alloc((size_t)DIM * DIM * 2);
  unsigned short* wvT  = (unsigned short*)alloc((size_t)DIM * DIM * 2);
  unsigned short* woT  = (unsigned short*)alloc((size_t)DIM * DIM * 2);
  unsigned short* adb  = (unsigned short*)alloc((size_t)ALEN * DIM * 2);
  unsigned short* Qbf  = (unsigned short*)alloc((size_t)MROWS * DIM * 2);
  unsigned short* Kbf  = (unsigned short*)alloc((size_t)MROWS * DIM * 2);
  unsigned short* Vbf  = (unsigned short*)alloc((size_t)MROWS * DIM * 2);
  unsigned short* VT   = (unsigned short*)alloc((size_t)MROWS * DIM * 2);
  unsigned short* akb  = (unsigned short*)alloc((size_t)ALEN * DIM * 2);
  unsigned short* avb  = (unsigned short*)alloc((size_t)ALEN * DIM * 2);
  float*          ctx  = (float*)alloc((size_t)MROWS * DIM * 4);
  unsigned short* ctxb = (unsigned short*)alloc((size_t)MROWS * DIM * 2);

  const long long nBig = (long long)MROWS * DIM;   // 16.7M
  cvt_f32_bf16<<<4096, 256, 0, stream>>>(x, xb, nBig);
  cvt_f32_bf16<<<64, 256, 0, stream>>>(adapter, adb, (long long)ALEN * DIM);

  dim3 gt(DIM / 32, DIM / 32);       // 128 x 128
  dim3 bt(32, 8);
  transpose_w<<<gt, bt, 0, stream>>>(wq, wqT);
  transpose_w<<<gt, bt, 0, stream>>>(wk, wkT);
  transpose_w<<<gt, bt, 0, stream>>>(wv, wvT);
  transpose_w<<<gt, bt, 0, stream>>>(wo, woT);

  dim3 gg(DIM / 128, MROWS / 128);   // 32 x 32
  gemm_bf16<false><<<gg, 256, 0, stream>>>(xb, wqT, nullptr, Qbf, MROWS, DIM, DIM);
  gemm_bf16<false><<<gg, 256, 0, stream>>>(xb, wkT, nullptr, Kbf, MROWS, DIM, DIM);
  gemm_bf16<false><<<gg, 256, 0, stream>>>(xb, wvT, nullptr, Vbf, MROWS, DIM, DIM);

  dim3 ga(DIM / 128, 1);
  gemm_bf16<false><<<ga, 256, 0, stream>>>(adb, wkT, nullptr, akb, ALEN, DIM, DIM);
  gemm_bf16<false><<<ga, 256, 0, stream>>>(adb, wvT, nullptr, avb, ALEN, DIM, DIM);

  long long nRope = (long long)MROWS * HEADS * (HD / 2);   // 8.4M
  rope_bf16<<<(unsigned)((nRope + 255) / 256), 256, 0, stream>>>(Qbf, fcos, fsin);
  rope_bf16<<<(unsigned)((nRope + 255) / 256), 256, 0, stream>>>(Kbf, fcos, fsin);

  dim3 gv(SEQ / 32, BSZ * HEADS * (HD / 32));  // 64 x 256
  transpose_v<<<gv, bt, 0, stream>>>(Vbf, VT);

  dim3 gat(SEQ / 128, BSZ * HEADS);  // 16 x 64
  attn_kernel<<<gat, 256, 0, stream>>>(Qbf, Kbf, VT, ctx);

  long long nAd = (long long)MROWS * HEADS;   // 131072
  adapter_kernel<<<(unsigned)((nAd + 255) / 256), 256, 0, stream>>>(Qbf, akb, avb, gate, ctx);

  cvt_f32_bf16<<<4096, 256, 0, stream>>>(ctx, ctxb, nBig);
  gemm_bf16<true><<<gg, 256, 0, stream>>>(ctxb, woT, (float*)d_out, nullptr, MROWS, DIM, DIM);
}
